// DWBNN_1151051235470
// MI455X (gfx1250) — matrix operand order repaired and checked by reading, not compiled, not run on hardware
//
#include <hip/hip_runtime.h>

#define B_SZ 4096
#define D0   2048
#define D1   8192
#define D2   2048
#define WC   8

typedef __attribute__((ext_vector_type(16))) __bf16 v16bf;
typedef __attribute__((ext_vector_type(8)))  float  v8f;

union Frag {
    v16bf v;
    uint4 q[2];
};

// ---------------------------------------------------------------------------
// 1) elementwise f32 -> bf16 (for x)
// ---------------------------------------------------------------------------
__global__ void f32_to_bf16_kernel(const float* __restrict__ in,
                                   __bf16* __restrict__ out, long n) {
    long i = ((long)blockIdx.x * blockDim.x + threadIdx.x) * 4;
    if (i + 3 < n) {
        float4 v = *(const float4*)(in + i);
        union { __bf16 h[4]; uint2 u; } p;
        p.h[0] = (__bf16)v.x; p.h[1] = (__bf16)v.y;
        p.h[2] = (__bf16)v.z; p.h[3] = (__bf16)v.w;
        *(uint2*)(out + i) = p.u;
    }
}

// ---------------------------------------------------------------------------
// 2) gather W_sel[i,o] = w[i,o,idx[i,o]], convert to bf16, store TRANSPOSED:
//    wt[o, i]  (so GEMM B-tiles are contiguous in K).
//    32x32 tile transposed through LDS: coalesced reads along o, coalesced
//    writes along i.
// ---------------------------------------------------------------------------
__global__ void gather_wt_kernel(const float* __restrict__ w,
                                 const int* __restrict__ idx,
                                 __bf16* __restrict__ wt,
                                 int Din, int Dout) {
    __shared__ __bf16 tile[32][34];   // 34*2B = 17 banks stride: conflict-free
    const int i0 = blockIdx.y * 32;
    const int o0 = blockIdx.x * 32;
    const int tx = threadIdx.x;       // 0..31 (along o on read, along i on write)
    const int ty = threadIdx.y;       // 0..7
#pragma unroll
    for (int r = 0; r < 4; ++r) {
        int il = ty + 8 * r;
        long p = (long)(i0 + il) * Dout + (o0 + tx);
        int c = idx[p];
        tile[il][tx] = (__bf16)w[p * WC + c];
    }
    __syncthreads();
#pragma unroll
    for (int r = 0; r < 4; ++r) {
        int ol = ty + 8 * r;
        wt[(long)(o0 + ol) * Din + (i0 + tx)] = tile[tx][ol];
    }
}

// ---------------------------------------------------------------------------
// 3) GEMM: out = relu(A[M,K] @ Bt[N,K]^T + bias[N])
//    bf16 inputs, f32 accumulate via v_wmma_f32_16x16x32_bf16.
//    Block: 128x128 tile, 256 threads = 8 waves (wave32).
//    Wave grid 2(M)x4(N): each wave -> 64x32 = 4x2 WMMA accumulators.
// ---------------------------------------------------------------------------
template <bool OUT_BF16>
__global__ __launch_bounds__(256)
void gemm_bias_relu_kernel(const __bf16* __restrict__ A,
                           const __bf16* __restrict__ Bt,
                           const float* __restrict__ bias,
                           void* __restrict__ outp,
                           int M, int N, int K) {
    // row stride 40 halves = 80B = 20 banks -> 16 half-wave lanes hit
    // distinct bank groups (20*i mod 64 all distinct, spacing 4).
    __shared__ __bf16 As[128 * 40];
    __shared__ __bf16 Bs[128 * 40];

    const int t     = threadIdx.x;
    const int lane  = t & 31;
    const int wave  = t >> 5;
    const int waveM = wave & 1;    // 2 slabs of 64 rows
    const int waveN = wave >> 1;   // 4 slabs of 32 cols
    const int hi    = lane >> 4;   // half-wave select
    const int l15   = lane & 15;

    const int m0 = blockIdx.y * 128;
    const int n0 = blockIdx.x * 128;

    v8f acc[4][2] = {};

    for (int k0 = 0; k0 < K; k0 += 32) {
        // --- stage 128x32 A-tile and 128x32 B-tile (Bt rows = N) into LDS ---
#pragma unroll
        for (int it = 0; it < 2; ++it) {
            int c    = t + 256 * it;       // 512 chunks of 8 halves per tile
            int row  = c >> 2;
            int colh = (c & 3) << 3;
            *(uint4*)&As[row * 40 + colh] =
                *(const uint4*)(A + (size_t)(m0 + row) * K + k0 + colh);
            *(uint4*)&Bs[row * 40 + colh] =
                *(const uint4*)(Bt + (size_t)(n0 + row) * K + k0 + colh);
        }
        __syncthreads();

        // --- fragments per CDNA5 WMMA VGPR layouts ---
        Frag a[4], b[2];
#pragma unroll
        for (int mt = 0; mt < 4; ++mt) {
            int row = waveM * 64 + mt * 16 + l15;     // A: lane holds row M=l15
            a[mt].q[0] = *(const uint4*)&As[row * 40 + 8 * hi];        // K 0..7 (+8)
            a[mt].q[1] = *(const uint4*)&As[row * 40 + 16 + 8 * hi];   // K 16..23 (+8)
        }
#pragma unroll
        for (int nt = 0; nt < 2; ++nt) {
            int row = waveN * 32 + nt * 16 + l15;     // B: lane holds col N=l15
            b[nt].q[0] = *(const uint4*)&Bs[row * 40 + 8 * hi];        // K 0..7 (+8)
            b[nt].q[1] = *(const uint4*)&Bs[row * 40 + 16 + 8 * hi];   // K 16..23 (+8)
        }

#pragma unroll
        for (int mt = 0; mt < 4; ++mt)
#pragma unroll
            for (int nt = 0; nt < 2; ++nt)
                acc[mt][nt] = __builtin_amdgcn_wmma_f32_16x16x32_bf16(
                    false, a[mt].v, false, b[nt].v,
                    (short)0, acc[mt][nt], false, false);
        __syncthreads();
    }

    // --- epilogue: bias + relu; C layout: n = l15, m = v + 8*hi ---
#pragma unroll
    for (int nt = 0; nt < 2; ++nt) {
        int n = n0 + waveN * 32 + nt * 16 + l15;
        float bv = bias[n];
#pragma unroll
        for (int mt = 0; mt < 4; ++mt) {
            int mbase = m0 + waveM * 64 + mt * 16 + 8 * hi;
#pragma unroll
            for (int v = 0; v < 8; ++v) {
                float r = acc[mt][nt][v] + bv;
                r = r > 0.0f ? r : 0.0f;
                size_t off = (size_t)(mbase + v) * N + n;
                if (OUT_BF16) ((__bf16*)outp)[off] = (__bf16)r;
                else          ((float*)outp)[off]  = r;
            }
        }
    }
}

// ---------------------------------------------------------------------------
extern "C" void kernel_launch(void* const* d_in, const int* in_sizes, int n_in,
                              void* d_out, int out_size, void* d_ws, size_t ws_size,
                              hipStream_t stream) {
    const float* x  = (const float*)d_in[0];
    const float* w1 = (const float*)d_in[1];
    const float* b1 = (const float*)d_in[2];
    const int*   i1 = (const int*)d_in[3];
    const float* w2 = (const float*)d_in[4];
    const float* b2 = (const float*)d_in[5];
    const int*   i2 = (const int*)d_in[6];

    // workspace layout (bytes): xb(16M) | w1t(32M) | hb(64M) | w2t(32M) = 144MB
    char* ws = (char*)d_ws;
    __bf16* xb  = (__bf16*)(ws);
    __bf16* w1t = (__bf16*)(ws + (size_t)B_SZ * D0 * 2);
    __bf16* hb  = (__bf16*)(ws + (size_t)B_SZ * D0 * 2 + (size_t)D1 * D0 * 2);
    __bf16* w2t = (__bf16*)(ws + (size_t)B_SZ * D0 * 2 + (size_t)D1 * D0 * 2 +
                                 (size_t)B_SZ * D1 * 2);

    // 1) x -> bf16
    long nx = (long)B_SZ * D0;
    f32_to_bf16_kernel<<<(int)(nx / (256 * 4)), 256, 0, stream>>>(x, xb, nx);

    // 2) gather+transpose selected weights to bf16
    gather_wt_kernel<<<dim3(D1 / 32, D0 / 32), dim3(32, 8), 0, stream>>>(
        w1, i1, w1t, D0, D1);
    gather_wt_kernel<<<dim3(D2 / 32, D1 / 32), dim3(32, 8), 0, stream>>>(
        w2, i2, w2t, D1, D2);

    // 3) layer 1: h = relu(x @ W1 + b1), h stored bf16
    gemm_bias_relu_kernel<true><<<dim3(D1 / 128, B_SZ / 128), 256, 0, stream>>>(
        xb, w1t, b1, (void*)hb, B_SZ, D1, D0);

    // 4) layer 2: out = relu(h @ W2 + b2), f32 out
    gemm_bias_relu_kernel<false><<<dim3(D2 / 128, B_SZ / 128), 256, 0, stream>>>(
        hb, w2t, b2, d_out, B_SZ, D2, D1);
}
